// Affine_Linear_Abla_Quat_44478681317765
// MI455X (gfx1250) — matrix operand order repaired and checked by reading, not compiled, not run on hardware
//
#include <hip/hip_runtime.h>
#include <stdint.h>

// ---------------------------------------------------------------------------
// Y[b,n,f,i] = sum_e A[f,e]*a[b,n,e,i] + B[f,e]*b[..] + C[f,e]*c[..]
// => batched GEMM per b: M=F=256, K=768 ([A|B|C]), Ncols=3*4096=12288.
// bf16 inputs, f32 accumulation via v_wmma_f32_16x16x32_bf16.
// Tile staging uses gfx1250 GLOBAL_LOAD_ASYNC_TO_LDS_B128 (ASYNCcnt) with
// double-buffered LDS so copies overlap WMMA compute.
// ---------------------------------------------------------------------------

typedef __attribute__((ext_vector_type(16))) __bf16 v16bf;
typedef __attribute__((ext_vector_type(8)))  __bf16 v8bf;
typedef __attribute__((ext_vector_type(8)))  float  v8f;

#define NB    8
#define NN    4096
#define DD    256
#define FF    256
#define KK    768              // 3*DD
#define NCOL  12288            // 3*NN
#define BM    128
#define BN    128
#define BK    32
#define LDSTR 80               // 32 bf16 (64B) + 16B pad -> conflict-free b128
#define TILEB (BM * LDSTR)     // 10240 bytes per W/T tile buffer
#define KITER (KK / BK)        // 24

__device__ __forceinline__ unsigned short f2bf(float f) {
    union { float f; unsigned u; } v; v.f = f;
    unsigned u = v.u;
    u += 0x7FFFu + ((u >> 16) & 1u);     // round-to-nearest-even
    return (unsigned short)(u >> 16);
}

// ---- Kernel 1: pack [A|B|C] f32 -> bf16 Wcat[256][768] ----------------------
__global__ void wcat_kernel(const float* __restrict__ A,
                            const float* __restrict__ B,
                            const float* __restrict__ C,
                            unsigned short* __restrict__ W) {
    int idx = blockIdx.x * 256 + threadIdx.x;      // 0 .. 196607
    int f = idx / KK, e = idx % KK;
    const float* src = (e < 256) ? A : (e < 512 ? B : C);
    W[idx] = f2bf(src[f * 256 + (e & 255)]);
}

// ---- Kernel 2: quaternion math -> bf16 terms T[b][3n+i][768] ----------------
__global__ __launch_bounds__(256)
void prep_kernel(const float* __restrict__ X, const float* __restrict__ J,
                 unsigned short* __restrict__ T) {
    const int d = threadIdx.x;                 // 0..255 (coalesced over d)
    const int n = blockIdx.x;
    const int b = blockIdx.y;
    const long idx = ((long)(b * NN + n)) * DD + d;

    const float4 q = ((const float4*)J)[idx];
    const float* xp = X + idx * 3;
    const float x0 = xp[0], x1 = xp[1], x2 = xp[2];

    float s  = q.x*q.x + q.y*q.y + q.z*q.z + q.w*q.w;
    float inv = 1.0f / fmaxf(sqrtf(s), 1e-12f);
    float x = q.x*inv, y = q.y*inv, z = q.z*inv, w = q.w*inv;
    float two_s = 2.0f / (w*w + x*x + y*y + z*z);

    float r00 = 1.0f - two_s*(y*y + z*z);
    float r01 = two_s*(x*y - z*w);
    float r02 = two_s*(x*z + y*w);
    float r10 = two_s*(x*y + z*w);
    float r11 = 1.0f - two_s*(x*x + z*z);
    float r12 = two_s*(y*z - x*w);
    float r20 = two_s*(x*z - y*w);
    float r21 = two_s*(y*z + x*w);
    float r22 = 1.0f - two_s*(x*x + y*y);

    float rt0 = r00*x0 + r10*x1 + r20*x2;
    float rt1 = r01*x0 + r11*x1 + r21*x2;
    float rt2 = r02*x0 + r12*x1 + r22*x2;

    float a0 = r00*rt0 + r01*rt1, a1 = r10*rt0 + r11*rt1, a2 = r20*rt0 + r21*rt1;
    float b0 = r01*rt0 - r00*rt1, b1 = r11*rt0 - r10*rt1, b2 = r21*rt0 - r20*rt1;
    float c0 = r02*rt2,           c1 = r12*rt2,           c2 = r22*rt2;

    unsigned long base = ((unsigned long)b * NCOL + (unsigned)(3*n)) * (unsigned long)KK;
    T[base +         d] = f2bf(a0);
    T[base + 256   + d] = f2bf(b0);
    T[base + 512   + d] = f2bf(c0);
    T[base + KK        + d] = f2bf(a1);
    T[base + KK + 256  + d] = f2bf(b1);
    T[base + KK + 512  + d] = f2bf(c1);
    T[base + 2*KK       + d] = f2bf(a2);
    T[base + 2*KK + 256 + d] = f2bf(b2);
    T[base + 2*KK + 512 + d] = f2bf(c2);
}

// Issue one tile's async copies: 128 rows x 32 bf16 for W and for T.
// 256 threads x 2 chunks x 16B per matrix. gW/gT already point at (row0, k0).
__device__ __forceinline__ void stage_tile_async(
        const unsigned short* gW, const unsigned short* gT,
        unsigned ldsW, unsigned ldsT, int tid) {
    #pragma unroll
    for (int j = 0; j < 2; ++j) {
        int ch = tid + j * 256;
        int r  = ch >> 2;                 // 0..127
        int qo = ch & 3;                  // 16B chunk within 64B row
        unsigned long long gaW = (unsigned long long)(gW + (unsigned)r * KK + qo * 8);
        unsigned           lW  = ldsW + (unsigned)(r * LDSTR + qo * 16);
        asm volatile("global_load_async_to_lds_b128 %0, %1, off"
                     :: "v"(lW), "v"(gaW) : "memory");
        unsigned long long gaT = (unsigned long long)(gT + (unsigned)r * KK + qo * 8);
        unsigned           lT  = ldsT + (unsigned)(r * LDSTR + qo * 16);
        asm volatile("global_load_async_to_lds_b128 %0, %1, off"
                     :: "v"(lT), "v"(gaT) : "memory");
    }
}

__device__ __forceinline__ void wait_async0() {
    asm volatile("s_wait_asynccnt 0" ::: "memory");
}

// ---- Kernel 3: bf16 GEMM with WMMA + async-to-LDS double buffering ----------
// grid: (NCOL/BN=96, FF/BM=2, NB=8), 256 threads = 8 waves (wave32)
__global__ __launch_bounds__(256)
void gemm_kernel(const unsigned short* __restrict__ W,   // [256][768] bf16
                 const unsigned short* __restrict__ T,   // [b][12288][768] bf16
                 float* __restrict__ Y) {
    __shared__ __align__(16) unsigned char sW[2][TILEB];  // 2 x 10 KB
    __shared__ __align__(16) unsigned char sT[2][TILEB];  // 2 x 10 KB

    const int tid   = threadIdx.x;
    const int lane  = tid & 31;
    const int wave  = tid >> 5;
    const int waveM = wave >> 2;          // 0..1  -> 64-row slab
    const int waveN = wave & 3;           // 0..3  -> 32-col slab
    const int lrow  = lane & 15;
    const int half  = lane >> 4;

    const int b  = blockIdx.z;
    const int f0 = blockIdx.y * BM;
    const int c0 = blockIdx.x * BN;

    const unsigned short* Wp = W + (unsigned)f0 * KK;
    const unsigned short* Tp = T + (unsigned long)b * (unsigned long)NCOL * KK
                                 + (unsigned long)c0 * KK;

    // LDS byte offsets (low 32 bits of generic pointer == LDS address)
    const unsigned sWoff = (unsigned)(unsigned long long)(void*)&sW[0][0];
    const unsigned sToff = (unsigned)(unsigned long long)(void*)&sT[0][0];

    v8f acc[4][2];
    #pragma unroll
    for (int mt = 0; mt < 4; ++mt)
        #pragma unroll
        for (int nt = 0; nt < 2; ++nt)
            acc[mt][nt] = (v8f){0.f,0.f,0.f,0.f,0.f,0.f,0.f,0.f};

    // Prologue: async-stage tile 0 into buffer 0
    stage_tile_async(Wp, Tp, sWoff, sToff, tid);

    for (int kt = 0; kt < KITER; ++kt) {
        const int buf = kt & 1;
        wait_async0();                    // this wave's tile-kt copies landed
        __syncthreads();                  // whole tile visible; prev buf free

        if (kt + 1 < KITER) {            // overlap: stage tile kt+1 into other buf
            const int k1 = (kt + 1) * BK;
            stage_tile_async(Wp + k1, Tp + k1,
                             sWoff + (unsigned)((buf ^ 1) * TILEB),
                             sToff + (unsigned)((buf ^ 1) * TILEB), tid);
        }

        const unsigned char* cW = &sW[buf][0];
        const unsigned char* cT = &sT[buf][0];

        // A fragments: 16x32 bf16; lanes 0-15 K{0-7,16-23}, lanes 16-31 K{8-15,24-31}
        v16bf afrag[4];
        #pragma unroll
        for (int mt = 0; mt < 4; ++mt) {
            int row = waveM * 64 + mt * 16 + lrow;
            v8bf lo = *(const v8bf*)(cW + row * LDSTR + half * 16);
            v8bf hi = *(const v8bf*)(cW + row * LDSTR + 32 + half * 16);
            afrag[mt] = __builtin_shufflevector(lo, hi, 0,1,2,3,4,5,6,7,8,9,10,11,12,13,14,15);
        }
        // B fragments: 32x16 bf16; lanes 0-15 K{0-7,16-23}, lanes 16-31 K{8-15,24-31}
        v16bf bfrag[2];
        #pragma unroll
        for (int nt = 0; nt < 2; ++nt) {
            int col = waveN * 32 + nt * 16 + lrow;
            v8bf lo = *(const v8bf*)(cT + col * LDSTR + half * 16);
            v8bf hi = *(const v8bf*)(cT + col * LDSTR + 32 + half * 16);
            bfrag[nt] = __builtin_shufflevector(lo, hi, 0,1,2,3,4,5,6,7,8,9,10,11,12,13,14,15);
        }

        #pragma unroll
        for (int mt = 0; mt < 4; ++mt)
            #pragma unroll
            for (int nt = 0; nt < 2; ++nt)
                acc[mt][nt] = __builtin_amdgcn_wmma_f32_16x16x32_bf16(
                    false, afrag[mt], false, bfrag[nt],
                    (short)0, acc[mt][nt], false, false);
    }

    // Epilogue: C/D layout -> VGPR v, lanes0-15: (M=v, N=lane); lanes16-31: (M=v+8)
    float* Yb = Y + (unsigned long)b * NN * (unsigned long)KK;   // b*4096*768
    #pragma unroll
    for (int nt = 0; nt < 2; ++nt) {
        int c = c0 + waveN * 32 + nt * 16 + lrow;                // column = 3n + i
        unsigned long colbase = (unsigned long)(c / 3) * KK + (unsigned)(c % 3);
        #pragma unroll
        for (int mt = 0; mt < 4; ++mt) {
            int fbase = f0 + waveM * 64 + mt * 16 + half * 8;
            #pragma unroll
            for (int vg = 0; vg < 8; ++vg)
                Yb[colbase + (unsigned)(fbase + vg) * 3u] = acc[mt][nt][vg];
        }
    }
}

// ---------------------------------------------------------------------------
extern "C" void kernel_launch(void* const* d_in, const int* in_sizes, int n_in,
                              void* d_out, int out_size, void* d_ws, size_t ws_size,
                              hipStream_t stream) {
    const float* X = (const float*)d_in[0];
    const float* J = (const float*)d_in[1];
    const float* A = (const float*)d_in[2];
    const float* B = (const float*)d_in[3];
    const float* C = (const float*)d_in[4];
    float* Y = (float*)d_out;

    unsigned short* Wcat = (unsigned short*)d_ws;                       // 384 KB
    unsigned short* T    = (unsigned short*)((char*)d_ws + 393216);     // 144 MB

    wcat_kernel<<<dim3((FF * KK) / 256), dim3(256), 0, stream>>>(A, B, C, Wcat);
    prep_kernel<<<dim3(NN, NB), dim3(256), 0, stream>>>(X, J, T);
    gemm_kernel<<<dim3(NCOL / BN, FF / BM, NB), dim3(256), 0, stream>>>(Wcat, T, Y);
}
